// ScheduledSamplingDecoder_7636451852789
// MI455X (gfx1250) — matrix operand order repaired and checked by reading, not compiled, not run on hardware
//
#include <hip/hip_runtime.h>
#include <hip/hip_bf16.h>

// Problem constants (match reference)
#define TT 128
#define BB 32
#define EE 512
#define HH 1024
#define VV 32000

typedef __attribute__((ext_vector_type(16))) __bf16 v16bf;
typedef __attribute__((ext_vector_type(8)))  float  v8f;

union FragU {
    v16bf v;
    uint4 q[2];
};

__device__ __forceinline__ unsigned short f2bf(float f) {
    unsigned int u = __float_as_uint(f);
    unsigned int r = u + 0x7FFFu + ((u >> 16) & 1u);   // round-to-nearest-even
    return (unsigned short)(r >> 16);
}

__device__ __forceinline__ float sigmoidf(float x) {
    return 1.0f / (1.0f + __expf(-x));
}

// CDNA5 async global -> LDS copy (tracked by ASYNCcnt), 16B per lane.
// lds_off: wave-relative LDS byte address (low 32 bits of a flat shared ptr).
__device__ __forceinline__ void async_g2l_b128(unsigned lds_off, const void* gptr) {
    asm volatile("global_load_async_to_lds_b128 %0, %1, off"
                 :: "v"(lds_off), "v"(gptr)
                 : "memory");
}

__device__ __forceinline__ void wait_asynccnt0() {
    asm volatile("s_wait_asynccnt 0" ::: "memory");
}

// ---------------------------------------------------------------------------
// fp32 -> bf16 conversion (grid-stride)
// ---------------------------------------------------------------------------
__global__ void f32_to_bf16_kernel(const float* __restrict__ src,
                                   unsigned short* __restrict__ dst, int n) {
    int i = blockIdx.x * blockDim.x + threadIdx.x;
    int stride = gridDim.x * blockDim.x;
    for (; i < n; i += stride) dst[i] = f2bf(src[i]);
}

// zero-init h0 (fp32 + bf16) — ws is poisoned, must re-zero each launch
__global__ void zero_h0_kernel(float* __restrict__ hf,
                               unsigned short* __restrict__ hbf, int n) {
    int i = blockIdx.x * blockDim.x + threadIdx.x;
    int stride = gridDim.x * blockDim.x;
    for (; i < n; i += stride) { hf[i] = 0.0f; hbf[i] = 0; }
}

// ---------------------------------------------------------------------------
// Embedding gather: x[t,b,:] = emb[inputs[max(t-1,0), b], :]  (fp32 -> bf16)
// ---------------------------------------------------------------------------
__global__ void embed_gather_kernel(const int* __restrict__ inputs,
                                    const float* __restrict__ emb,
                                    unsigned short* __restrict__ x_bf) {
    const int row = blockIdx.x;             // t*B + b
    const int t = row / BB;
    const int b = row - t * BB;
    const int tsrc = (t == 0) ? 0 : (t - 1);
    const int idx = inputs[tsrc * BB + b];
    const float* __restrict__ src = emb + (size_t)idx * EE;
    unsigned short* __restrict__ dst = x_bf + (size_t)row * EE;
    for (int i = threadIdx.x; i < EE; i += blockDim.x) dst[i] = f2bf(src[i]);
}

// ---------------------------------------------------------------------------
// bf16 WMMA GEMM with double-buffered async global->LDS staging:
//   C[M,N] = A[M,K] * B[N,K]^T + bias[N]   (fp32 out)
// Block: 256 threads = 8 waves, block tile 64x256, wave tile 32x64, K-step 32.
// Per K-step each thread issues 5 global_load_async_to_lds_b128 (ASYNCcnt),
// pipeline: wait_asynccnt 0 -> barrier -> stage next -> ds_load frags -> WMMA.
// Requires M%64==0, N%256==0, K%32==0.
// ---------------------------------------------------------------------------
#define LDS_STRIDE 40   // 32 K-elements + 8 pad (80B rows: conflict-free b128)

__global__ __launch_bounds__(256)
void gemm_bf16_f32_kernel(const unsigned short* __restrict__ A,
                          const unsigned short* __restrict__ B,
                          const float* __restrict__ bias,
                          float* __restrict__ C,
                          int M, int N, int K) {
    __shared__ unsigned short Abuf[2][64 * LDS_STRIDE];    // 2 x  5120 B
    __shared__ unsigned short Bbuf[2][256 * LDS_STRIDE];   // 2 x 20480 B

    const int tid  = threadIdx.x;
    const int lane = tid & 31;
    const int wave = tid >> 5;
    const int wm = wave >> 2;               // 0..1
    const int wn = wave & 3;                // 0..3
    const int lr = lane & 15;
    const int lh = lane >> 4;

    const int m0b = blockIdx.y * 64;        // block M origin
    const int n0b = blockIdx.x * 256;       // block N origin

    // Wave-relative LDS byte offsets of the staging buffers
    const unsigned aBase[2] = { (unsigned)(size_t)(&Abuf[0][0]),
                                (unsigned)(size_t)(&Abuf[1][0]) };
    const unsigned bBase[2] = { (unsigned)(size_t)(&Bbuf[0][0]),
                                (unsigned)(size_t)(&Bbuf[1][0]) };

    // Stage one 64x32 A-slice + 256x32 B-slice at k=kk into buffer s.
    auto stage = [&](int kk, int s) {
        {   // A tile: 256 lane-copies of 16B -> 1 per thread
            const int row = tid >> 2;
            const int ch  = (tid & 3) * 8;
            const unsigned short* g = A + (size_t)(m0b + row) * K + kk + ch;
            async_g2l_b128(aBase[s] + (unsigned)(row * LDS_STRIDE + ch) * 2u, g);
        }
#pragma unroll
        for (int c = 0; c < 4; ++c) {       // B tile: 4 per thread
            const unsigned short* g = B + (size_t)(n0b + tid) * K + kk + c * 8;
            async_g2l_b128(bBase[s] + (unsigned)(tid * LDS_STRIDE + c * 8) * 2u, g);
        }
    };

    v8f acc[2][4];
#pragma unroll
    for (int mi = 0; mi < 2; ++mi)
#pragma unroll
        for (int ni = 0; ni < 4; ++ni) acc[mi][ni] = (v8f){0,0,0,0,0,0,0,0};

    int cur = 0;
    stage(0, 0);

    for (int kk = 0; kk < K; kk += 32) {
        wait_asynccnt0();       // my stage of buffer `cur` complete
        __syncthreads();        // everyone's stage complete; prev reads done
        if (kk + 32 < K) stage(kk + 32, cur ^ 1);

        v16bf afrag[2];
#pragma unroll
        for (int mi = 0; mi < 2; ++mi) {
            const unsigned short* p =
                &Abuf[cur][(wm * 32 + mi * 16 + lr) * LDS_STRIDE + lh * 8];
            FragU f;
            f.q[0] = *(const uint4*)(p);
            f.q[1] = *(const uint4*)(p + 16);
            afrag[mi] = f.v;
        }
        v16bf bfrag[4];
#pragma unroll
        for (int ni = 0; ni < 4; ++ni) {
            const unsigned short* p =
                &Bbuf[cur][(wn * 64 + ni * 16 + lr) * LDS_STRIDE + lh * 8];
            FragU f;
            f.q[0] = *(const uint4*)(p);
            f.q[1] = *(const uint4*)(p + 16);
            bfrag[ni] = f.v;
        }
#pragma unroll
        for (int mi = 0; mi < 2; ++mi)
#pragma unroll
            for (int ni = 0; ni < 4; ++ni)
                acc[mi][ni] = __builtin_amdgcn_wmma_f32_16x16x32_bf16(
                    false, afrag[mi], false, bfrag[ni],
                    (short)0, acc[mi][ni], false, false);
        cur ^= 1;
    }

#pragma unroll
    for (int ni = 0; ni < 4; ++ni) {
        const int col = n0b + wn * 64 + ni * 16 + lr;
        const float bv = bias ? bias[col] : 0.0f;
#pragma unroll
        for (int mi = 0; mi < 2; ++mi) {
#pragma unroll
            for (int r = 0; r < 8; ++r) {
                const int row = m0b + wm * 32 + mi * 16 + r + lh * 8;
                C[(size_t)row * N + col] = acc[mi][ni][r] + bv;
            }
        }
    }
}

// ---------------------------------------------------------------------------
// One GRU step, fully fused (direct-global; w_hh is L2-resident):
//   gh = h_{t-1} @ w_hh^T  (WMMA bf16 -> fp32), then gate math in fp32.
// Each wave owns one 16-wide H-column tile and all 3 gate tiles
// (6 accumulators: 3 gates x 2 row-tiles of the 32-row batch).
// Grid: 8 blocks x 256 threads = 64 waves = H/16 column tiles.
// ---------------------------------------------------------------------------
__global__ __launch_bounds__(256)
void gru_step_kernel(const unsigned short* __restrict__ hprev_bf, // [B,H] bf16
                     const float* __restrict__ hprev_f,           // [B,H] f32
                     const unsigned short* __restrict__ whh_bf,   // [3H,H] bf16
                     const float* __restrict__ b_hh,              // [3H]
                     const float* __restrict__ gx_t,              // [B,3H] f32 (incl. b_ih)
                     unsigned short* __restrict__ hnew_bf,        // [B,H] bf16
                     float* __restrict__ hnew_f,                  // [B,H] f32
                     float* __restrict__ hlast) {                 // [B,H] f32 or null
    const int lane = threadIdx.x & 31;
    const int wave = threadIdx.x >> 5;
    const int jt = blockIdx.x * 8 + wave;   // 0..63 column tile
    const int lr = lane & 15;
    const int lh = lane >> 4;

    v8f acc[3][2];
#pragma unroll
    for (int g = 0; g < 3; ++g)
#pragma unroll
        for (int mi = 0; mi < 2; ++mi) acc[g][mi] = (v8f){0,0,0,0,0,0,0,0};

    for (int kk = 0; kk < HH; kk += 32) {
        const int ka = kk + lh * 8;
        const int kb = kk + lh * 8;

        v16bf afrag[2];
#pragma unroll
        for (int mi = 0; mi < 2; ++mi) {
            const unsigned short* p = hprev_bf + (size_t)(mi * 16 + lr) * HH + ka;
            FragU f;
            f.q[0] = *(const uint4*)(p);
            f.q[1] = *(const uint4*)(p + 16);
            afrag[mi] = f.v;
        }
        v16bf bfrag[3];
#pragma unroll
        for (int g = 0; g < 3; ++g) {
            const unsigned short* p =
                whh_bf + (size_t)(g * HH + jt * 16 + lr) * HH + kb;
            FragU f;
            f.q[0] = *(const uint4*)(p);
            f.q[1] = *(const uint4*)(p + 16);
            bfrag[g] = f.v;
        }
#pragma unroll
        for (int g = 0; g < 3; ++g)
#pragma unroll
            for (int mi = 0; mi < 2; ++mi)
                acc[g][mi] = __builtin_amdgcn_wmma_f32_16x16x32_bf16(
                    false, afrag[mi], false, bfrag[g],
                    (short)0, acc[g][mi], false, false);
    }

    const int col = jt * 16 + lr;
    const float bhr = b_hh[col];
    const float bhz = b_hh[HH + col];
    const float bhn = b_hh[2 * HH + col];

#pragma unroll
    for (int mi = 0; mi < 2; ++mi) {
#pragma unroll
        for (int r = 0; r < 8; ++r) {
            const int b = mi * 16 + r + lh * 8;      // batch row 0..31
            const size_t gxbase = (size_t)b * (3 * HH);
            const float xr = gx_t[gxbase + col];
            const float xz = gx_t[gxbase + HH + col];
            const float xn = gx_t[gxbase + 2 * HH + col];
            const float hr = acc[0][mi][r] + bhr;
            const float hz = acc[1][mi][r] + bhz;
            const float hn = acc[2][mi][r] + bhn;
            const float rg = sigmoidf(xr + hr);
            const float zg = sigmoidf(xz + hz);
            const float ng = tanhf(xn + rg * hn);
            const float hp = hprev_f[(size_t)b * HH + col];
            const float hv = (1.0f - zg) * ng + zg * hp;
            hnew_f[(size_t)b * HH + col] = hv;
            hnew_bf[(size_t)b * HH + col] = f2bf(hv);
            if (hlast) hlast[(size_t)b * HH + col] = hv;
        }
    }
}

// ---------------------------------------------------------------------------
// Host-side orchestration
// ---------------------------------------------------------------------------
static inline size_t align256(size_t x) { return (x + 255) & ~(size_t)255; }

extern "C" void kernel_launch(void* const* d_in, const int* in_sizes, int n_in,
                              void* d_out, int out_size, void* d_ws, size_t ws_size,
                              hipStream_t stream) {
    // Inputs (setup_inputs order)
    const int*   inputs = (const int*)d_in[0];
    // d_in[1] encoder_output, d_in[2] encoder_mask: unused by reference math
    const float* emb    = (const float*)d_in[3];   // [V,E]
    const float* w_ih   = (const float*)d_in[4];   // [3H,E]
    const float* w_hh   = (const float*)d_in[5];   // [3H,H]
    const float* b_ih   = (const float*)d_in[6];   // [3H]
    const float* b_hh   = (const float*)d_in[7];   // [3H]
    const float* out_w  = (const float*)d_in[8];   // [V,H]
    const float* out_b  = (const float*)d_in[9];   // [V]

    float* out = (float*)d_out;                    // logits [T,B,V] ++ h_last [B,H]

    // Workspace carve-up
    char* ws = (char*)d_ws;
    size_t off = 0;
    unsigned short* w_ih_bf  = (unsigned short*)(ws + off); off = align256(off + sizeof(unsigned short) * (size_t)3 * HH * EE);
    unsigned short* w_hh_bf  = (unsigned short*)(ws + off); off = align256(off + sizeof(unsigned short) * (size_t)3 * HH * HH);
    unsigned short* out_w_bf = (unsigned short*)(ws + off); off = align256(off + sizeof(unsigned short) * (size_t)VV * HH);
    unsigned short* x_bf     = (unsigned short*)(ws + off); off = align256(off + sizeof(unsigned short) * (size_t)TT * BB * EE);
    unsigned short* hall_bf  = (unsigned short*)(ws + off); off = align256(off + sizeof(unsigned short) * (size_t)TT * BB * HH);
    unsigned short* h0_bf    = (unsigned short*)(ws + off); off = align256(off + sizeof(unsigned short) * (size_t)BB * HH);
    float* gx      = (float*)(ws + off); off = align256(off + sizeof(float) * (size_t)TT * BB * 3 * HH);
    float* hall_f  = (float*)(ws + off); off = align256(off + sizeof(float) * (size_t)TT * BB * HH);
    float* h0_f    = (float*)(ws + off); off = align256(off + sizeof(float) * (size_t)BB * HH);
    (void)ws_size; (void)n_in; (void)in_sizes; (void)out_size;

    // 1) Convert weights to bf16
    f32_to_bf16_kernel<<<2048, 256, 0, stream>>>(w_ih,  w_ih_bf,  3 * HH * EE);
    f32_to_bf16_kernel<<<2048, 256, 0, stream>>>(w_hh,  w_hh_bf,  3 * HH * HH);
    f32_to_bf16_kernel<<<4096, 256, 0, stream>>>(out_w, out_w_bf, VV * HH);

    // 2) Zero h0 (ws is poisoned / dirty from previous replay)
    zero_h0_kernel<<<128, 256, 0, stream>>>(h0_f, h0_bf, BB * HH);

    // 3) Embedding gather with the teacher-forcing shift
    embed_gather_kernel<<<TT * BB, 128, 0, stream>>>(inputs, emb, x_bf);

    // 4) gx = x @ w_ih^T + b_ih   : M=4096, N=3072, K=512
    {
        dim3 grid((3 * HH) / 256, (TT * BB) / 64);
        gemm_bf16_f32_kernel<<<grid, 256, 0, stream>>>(
            x_bf, w_ih_bf, b_ih, gx, TT * BB, 3 * HH, EE);
    }

    // 5) Sequential GRU scan (stream-ordered launches give the T dependency)
    for (int t = 0; t < TT; ++t) {
        const unsigned short* hp_bf = (t == 0) ? h0_bf : (hall_bf + (size_t)(t - 1) * BB * HH);
        const float*          hp_f  = (t == 0) ? h0_f  : (hall_f  + (size_t)(t - 1) * BB * HH);
        float* hlast = (t == TT - 1) ? (out + (size_t)TT * BB * VV) : nullptr;
        gru_step_kernel<<<8, 256, 0, stream>>>(
            hp_bf, hp_f, w_hh_bf, b_hh,
            gx + (size_t)t * BB * 3 * HH,
            hall_bf + (size_t)t * BB * HH,
            hall_f  + (size_t)t * BB * HH,
            hlast);
    }

    // 6) logits = h_all @ out_w^T + out_b : M=4096, N=32000, K=1024
    {
        dim3 grid(VV / 256, (TT * BB) / 64);
        gemm_bf16_f32_kernel<<<grid, 256, 0, stream>>>(
            hall_bf, out_w_bf, out_b, out, TT * BB, VV, HH);
    }
}
